// TimeAwareGAT_77653008712124
// MI455X (gfx1250) — matrix operand order repaired and checked by reading, not compiled, not run on hardware
//
#include <hip/hip_runtime.h>
#include <hip/hip_bf16.h>

#define NN    100000   // nodes
#define RR    500      // relations
#define HIDD  128
#define NHEAD 8
#define EE    600000   // edges
#define NEIDN 100000
#define SLOPE 0.2f

typedef __attribute__((ext_vector_type(16))) __bf16 v16bf;
typedef __attribute__((ext_vector_type(2)))  __bf16 v2bf;
typedef __attribute__((ext_vector_type(8)))  float  v8f;

__device__ __forceinline__ unsigned short f2bfu(float f) {
    unsigned u = __float_as_uint(f);
    u += 0x7FFFu + ((u >> 16) & 1u);           // round-to-nearest-even
    return (unsigned short)(u >> 16);
}
__device__ __forceinline__ __bf16 f2bf(float f) {
    unsigned short s = f2bfu(f);
    return __builtin_bit_cast(__bf16, s);
}
// packed f32x2 -> bf16x2 (hardware cvt when the builtin exists)
__device__ __forceinline__ v2bf f2bf2(float a, float b) {
#if __has_builtin(__builtin_amdgcn_cvt_pk_bf16_f32)
    return __builtin_amdgcn_cvt_pk_bf16_f32(a, b);
#else
    v2bf r;
    r[0] = f2bf(a);
    r[1] = f2bf(b);
    return r;
#endif
}
__device__ __forceinline__ float lrelu(float x) { return x > 0.f ? x : SLOPE * x; }

__device__ __forceinline__ void atomicMaxF(float* addr, float val) {
    int cur = __float_as_int(*addr);
    while (val > __int_as_float(cur)) {
        int prev = atomicCAS((int*)addr, cur, __float_as_int(val));
        if (prev == cur) break;
        cur = prev;
    }
}

// ---------------------------------------------------------------------------
// Fold attention vectors into the projection weights.
//  block b: layer = b/152, col = b%152 ; 128 threads = input dim k.
//  col <128   : Wcat^T[col][k] = lin[k][col]                (bf16)
//  col 128-135: M_s column h  = sum_c lin[k][h*16+c]*a_s    (bf16)
//  col 136-143: M_d column h                                (bf16)
//  col 144-151: M_e[k][h]     = sum_c lin_e[k][h*16+c]*a_e  (f32, kept)
// ---------------------------------------------------------------------------
__global__ void prep_wcat_kernel(const float* __restrict__ g1_lin, const float* __restrict__ g1_lin_e,
                                 const float* __restrict__ g1_as,  const float* __restrict__ g1_ad,
                                 const float* __restrict__ g1_ae,
                                 const float* __restrict__ g2_lin, const float* __restrict__ g2_lin_e,
                                 const float* __restrict__ g2_as,  const float* __restrict__ g2_ad,
                                 const float* __restrict__ g2_ae,
                                 unsigned short* __restrict__ wcatT, float* __restrict__ Me) {
    int b = blockIdx.x;
    int layer = b / 152;
    int col   = b % 152;
    int k = threadIdx.x;  // 0..127
    const float* lin   = layer ? g2_lin   : g1_lin;
    const float* lin_e = layer ? g2_lin_e : g1_lin_e;
    const float* avs   = layer ? g2_as : g1_as;
    const float* avd   = layer ? g2_ad : g1_ad;
    const float* ave   = layer ? g2_ae : g1_ae;
    unsigned short* wt = wcatT + (size_t)layer * 144 * 128;
    float* me = Me + (size_t)layer * 128 * 8;
    if (col < 128) {
        wt[col * 128 + k] = f2bfu(lin[k * 128 + col]);
    } else if (col < 136) {
        int h = col - 128; float acc = 0.f;
        for (int c = 0; c < 16; ++c) acc += lin[k * 128 + h * 16 + c] * avs[h * 16 + c];
        wt[col * 128 + k] = f2bfu(acc);
    } else if (col < 144) {
        int h = col - 136; float acc = 0.f;
        for (int c = 0; c < 16; ++c) acc += lin[k * 128 + h * 16 + c] * avd[h * 16 + c];
        wt[col * 128 + k] = f2bfu(acc);
    } else {
        int h = col - 144; float acc = 0.f;
        for (int c = 0; c < 16; ++c) acc += lin_e[k * 128 + h * 16 + c] * ave[h * 16 + c];
        me[k * 8 + h] = acc;
    }
}

// rel_al[L][r][h] = rel_w[r]·Me_L[:,h]; TW[L][t][h] = tp_w[t]·Me_L[:,h]; tb[L][h] = tp_b·Me_L[:,h]
__global__ void rel_tw_kernel(const float* __restrict__ rel_w, const float* __restrict__ tp_w,
                              const float* __restrict__ tp_b,  const float* __restrict__ Me,
                              float* __restrict__ relAl, float* __restrict__ TW, float* __restrict__ tb) {
    int tid = blockIdx.x * blockDim.x + threadIdx.x;
    if (tid >= 2 * 4520) return;           // 2 layers * (500 + 64 + 1) * 8
    int layer = tid / 4520;
    int rem   = tid % 4520;
    int r = rem / 8, h = rem % 8;
    const float* me = Me + layer * 1024;
    float acc = 0.f;
    if (r < 500) {
        for (int k = 0; k < 128; ++k) acc += rel_w[r * 128 + k] * me[k * 8 + h];
        relAl[layer * 4000 + r * 8 + h] = acc;
    } else if (r < 564) {
        int t = r - 500;
        for (int k = 0; k < 128; ++k) acc += tp_w[t * 128 + k] * me[k * 8 + h];
        TW[layer * 512 + t * 8 + h] = acc;
    } else {
        for (int k = 0; k < 128; ++k) acc += tp_b[k] * me[k * 8 + h];
        tb[layer * 8 + h] = acc;
    }
}

// Per edge: al_e for both layers via the folded rank-8 projection; also reduce
// feature means + rel means (needed for the self-loop edge attribute).
__global__ __launch_bounds__(256)
void edge_al_kernel(const float* __restrict__ edge_t, const int* __restrict__ edge_type,
                    const float* __restrict__ relAl, const float* __restrict__ TW,
                    const float* __restrict__ tb, float* __restrict__ alE,
                    float* __restrict__ gsums) {
    __shared__ float sTW[1024];
    __shared__ float sTB[16];
    for (int i = threadIdx.x; i < 1024; i += 256) sTW[i] = TW[i];
    if (threadIdx.x < 16) sTB[threadIdx.x] = tb[threadIdx.x];
    __syncthreads();

    int e = blockIdx.x * 256 + threadIdx.x;
    bool active = (e < EE);
    int ee = active ? e : 0;
    float t0 = edge_t[(size_t)ee * 2 + 0];
    float t1 = edge_t[(size_t)ee * 2 + 1];
    int typ = edge_type[ee];

    float f[64];
#pragma unroll
    for (int j = 0; j < 16; ++j) {
        float invf = __expf(-(float)j * 0.57564627324851142f);  // ln(10000)/16
        float s0, c0, s1, c1;
        __sincosf(t0 * invf, &s0, &c0);
        __sincosf(t1 * invf, &s1, &c1);
        f[j] = s0; f[16 + j] = c0; f[32 + j] = s1; f[48 + j] = c1;
    }
    if (!active) {
#pragma unroll
        for (int j = 0; j < 64; ++j) f[j] = 0.f;
    }

    float a1[8], a2[8];
#pragma unroll
    for (int h = 0; h < 8; ++h) { a1[h] = sTB[h]; a2[h] = sTB[8 + h]; }
#pragma unroll
    for (int t = 0; t < 64; ++t) {
        float ft = f[t];
        const float* w1 = &sTW[t * 8];
        const float* w2 = &sTW[512 + t * 8];
#pragma unroll
        for (int h = 0; h < 8; ++h) { a1[h] += ft * w1[h]; a2[h] += ft * w2[h]; }
    }

    float r1[8], r2[8];
#pragma unroll
    for (int h = 0; h < 8; ++h) {
        r1[h] = relAl[typ * 8 + h];
        r2[h] = relAl[4000 + typ * 8 + h];
    }
    if (active) {
#pragma unroll
        for (int h = 0; h < 8; ++h) {
            alE[(size_t)e * 8 + h]                    = r1[h] + a1[h];
            alE[(size_t)EE * 8 + (size_t)e * 8 + h]   = r2[h] + a2[h];
        }
    } else {
#pragma unroll
        for (int h = 0; h < 8; ++h) { r1[h] = 0.f; r2[h] = 0.f; }
    }

    // wave reduction for the means
    int lane = threadIdx.x & 31;
#pragma unroll
    for (int t = 0; t < 64; ++t) {
        float v = f[t];
        for (int off = 16; off > 0; off >>= 1) v += __shfl_down(v, off);
        if (lane == 0) atomicAdd(&gsums[t], v);
    }
#pragma unroll
    for (int h = 0; h < 8; ++h) {
        float v = r1[h];
        for (int off = 16; off > 0; off >>= 1) v += __shfl_down(v, off);
        if (lane == 0) atomicAdd(&gsums[64 + h], v);
        float w = r2[h];
        for (int off = 16; off > 0; off >>= 1) w += __shfl_down(w, off);
        if (lane == 0) atomicAdd(&gsums[72 + h], w);
    }
}

// self-loop attention logits: loopAl[L][h] = mean(rel_al) + mean(feats)·TW + tb
__global__ void loop_al_kernel(const float* __restrict__ gsums, const float* __restrict__ TW,
                               const float* __restrict__ tb, float* __restrict__ loopAl) {
    int tid = threadIdx.x;
    if (tid >= 16) return;
    int layer = tid >> 3, h = tid & 7;
    float inv = 1.0f / (float)EE;
    float acc = gsums[64 + layer * 8 + h] * inv + tb[layer * 8 + h];
    for (int t = 0; t < 64; ++t) acc += (gsums[t] * inv) * TW[layer * 512 + t * 8 + h];
    loopAl[tid] = acc;
}

// ---------------------------------------------------------------------------
// WMMA GEMM: [128 rows/block] x = [N,128]f32 -> xp_cat = x @ WcatT^T [N,144]
// 8 waves, each 16 rows x 9 column-tiles; K=128 in 4 chunks of 32 (bf16).
// Tiles 0..7 -> xp[N,128]; tile 8 cols 0..7 -> al_s, 8..15 -> al_d.
// ---------------------------------------------------------------------------
__global__ __launch_bounds__(256)
void gemm_xp_kernel(const float* __restrict__ x, const unsigned short* __restrict__ wcatT,
                    float* __restrict__ xp, float* __restrict__ als, float* __restrict__ ald) {
    __shared__ __align__(16) unsigned short Bs[144 * 128];  // [col][k], bf16, 36 KB
    {
        const uint4* src = (const uint4*)wcatT;
        uint4* dst = (uint4*)Bs;
        for (int i = threadIdx.x; i < 144 * 128 / 8; i += 256) dst[i] = src[i];
    }
    __syncthreads();

    int wave = threadIdx.x >> 5, lane = threadIdx.x & 31;
    int g = lane >> 4, lr = lane & 15;
    int row0 = blockIdx.x * 128 + wave * 16;
    int arow = row0 + lr;
    if (arow >= NN) arow = NN - 1;
    const float* xr = x + (size_t)arow * HIDD;

    v8f acc[9];
#pragma unroll
    for (int t = 0; t < 9; ++t) acc[t] = {};

#pragma unroll
    for (int kk = 0; kk < 4; ++kk) {
        // A fragment: bf16 16x32, documented VGPR layout
        v16bf afrag;
#pragma unroll
        for (int j = 0; j < 8; ++j) {
            int kb = kk * 32 + ((j < 4) ? (8 * g + 2 * j) : (16 + 8 * g + 2 * (j - 4)));
            float2 p = *(const float2*)(xr + kb);
            v2bf pr = f2bf2(p.x, p.y);
            afrag[2 * j]     = pr[0];
            afrag[2 * j + 1] = pr[1];
        }
#pragma unroll
        for (int t = 0; t < 9; ++t) {
            v16bf bfrag;
            const unsigned short* bcol = &Bs[(t * 16 + lr) * 128 + kk * 32 + 8 * g];
#pragma unroll
            for (int j = 0; j < 8; ++j) {
                unsigned int pv = *(const unsigned int*)(bcol + ((j < 4) ? 2 * j : 16 + 2 * (j - 4)));
                unsigned short lo = (unsigned short)(pv & 0xFFFFu);
                unsigned short hi = (unsigned short)(pv >> 16);
                bfrag[2 * j]     = __builtin_bit_cast(__bf16, lo);
                bfrag[2 * j + 1] = __builtin_bit_cast(__bf16, hi);
            }
            acc[t] = __builtin_amdgcn_wmma_f32_16x16x32_bf16(
                false, afrag, false, bfrag, (short)0, acc[t], false, false);
        }
    }

    // C layout: col = lane&15, rows 8*g + j
#pragma unroll
    for (int j = 0; j < 8; ++j) {
        int r = row0 + 8 * g + j;
        if (r < NN) {
#pragma unroll
            for (int t = 0; t < 8; ++t) xp[(size_t)r * HIDD + t * 16 + lr] = acc[t][j];
            if (lr < 8) als[r * 8 + lr]       = acc[8][j];
            else        ald[r * 8 + (lr - 8)] = acc[8][j];
        }
    }
}

// seed segment-max with the self-loop logit (every node has one)
__global__ void node_init_kernel(const float* __restrict__ als, const float* __restrict__ ald,
                                 const float* __restrict__ loopAl, float* __restrict__ amax) {
    int n = blockIdx.x * blockDim.x + threadIdx.x;
    if (n >= NN) return;
#pragma unroll
    for (int h = 0; h < 8; ++h)
        amax[n * 8 + h] = lrelu(als[n * 8 + h] + ald[n * 8 + h] + loopAl[h]);
}

__global__ void edge_max_kernel(const int* __restrict__ ei, const float* __restrict__ als,
                                const float* __restrict__ ald, const float* __restrict__ alE,
                                float* __restrict__ amax) {
    int e = blockIdx.x * blockDim.x + threadIdx.x;
    if (e >= EE) return;
    int s = ei[e], d = ei[EE + e];
#pragma unroll
    for (int h = 0; h < 8; ++h) {
        float a = lrelu(als[s * 8 + h] + ald[d * 8 + h] + alE[(size_t)e * 8 + h]);
        atomicMaxF(&amax[d * 8 + h], a);
    }
}

// seed denom/num with the self-loop contribution (no atomics needed)
__global__ void node_start_kernel(const float* __restrict__ als, const float* __restrict__ ald,
                                  const float* __restrict__ loopAl, const float* __restrict__ amax,
                                  const float* __restrict__ xp, float* __restrict__ denom,
                                  float* __restrict__ num) {
    int n = blockIdx.x * blockDim.x + threadIdx.x;
    if (n >= NN) return;
#pragma unroll
    for (int h = 0; h < 8; ++h) {
        float aL = lrelu(als[n * 8 + h] + ald[n * 8 + h] + loopAl[h]);
        float w = __expf(aL - amax[n * 8 + h]);
        denom[n * 8 + h] = w;
        const float4* xv = (const float4*)&xp[(size_t)n * HIDD + h * 16];
        float4* nv = (float4*)&num[(size_t)n * HIDD + h * 16];
#pragma unroll
        for (int c = 0; c < 4; ++c) {
            float4 v = xv[c];
            nv[c] = make_float4(w * v.x, w * v.y, w * v.z, w * v.w);
        }
    }
}

// 32 lanes per edge, each lane scatters 4 channels
__global__ __launch_bounds__(256)
void edge_accum_kernel(const int* __restrict__ ei, const float* __restrict__ als,
                       const float* __restrict__ ald, const float* __restrict__ alE,
                       const float* __restrict__ amax, const float* __restrict__ xp,
                       float* __restrict__ denom, float* __restrict__ num) {
    long long gid = (long long)blockIdx.x * 256 + threadIdx.x;
    if (gid >= (long long)EE * 32) return;
    int e = (int)(gid >> 5);
    int j = (int)(gid & 31);
    int s = ei[e], d = ei[EE + e];
    int h = j >> 2;
    float a = lrelu(als[s * 8 + h] + ald[d * 8 + h] + alE[(size_t)e * 8 + h]);
    float w = __expf(a - amax[d * 8 + h]);
    if ((j & 3) == 0) atomicAdd(&denom[d * 8 + h], w);
    float4 v = *(const float4*)&xp[(size_t)s * HIDD + j * 4];
    float* np = &num[(size_t)d * HIDD + j * 4];
    atomicAdd(np + 0, w * v.x);
    atomicAdd(np + 1, w * v.y);
    atomicAdd(np + 2, w * v.z);
    atomicAdd(np + 3, w * v.w);
}

__global__ void node_fin_kernel(const float* __restrict__ num, const float* __restrict__ denom,
                                const float* __restrict__ bias, const float* __restrict__ prev,
                                float* __restrict__ out) {
    int n = blockIdx.x * blockDim.x + threadIdx.x;
    if (n >= NN) return;
#pragma unroll
    for (int h = 0; h < 8; ++h) {
        float inv = 1.0f / (denom[n * 8 + h] + 1e-16f);
#pragma unroll
        for (int c = 0; c < 16; ++c) {
            int k = h * 16 + c;
            float v = num[(size_t)n * HIDD + k] * inv + bias[k];
            if (prev) v += prev[(size_t)n * HIDD + k];
            out[(size_t)n * HIDD + k] = v;
        }
    }
}

__global__ void out_kernel(const int* __restrict__ ei, const int* __restrict__ eids,
                           const float* __restrict__ h, const float* __restrict__ out_w,
                           const float* __restrict__ out_b, float* __restrict__ pred) {
    int i = blockIdx.x * blockDim.x + threadIdx.x;
    if (i >= NEIDN) return;
    int e = eids[i];
    int s = ei[e], d = ei[EE + e];
    float o0 = out_b[0], o1 = out_b[1];
    const float4* hs = (const float4*)&h[(size_t)s * HIDD];
    const float4* hd = (const float4*)&h[(size_t)d * HIDD];
#pragma unroll 4
    for (int q = 0; q < 32; ++q) {
        float4 a = hs[q], b = hd[q];
        float4 he = make_float4(0.5f * (a.x + b.x), 0.5f * (a.y + b.y),
                                0.5f * (a.z + b.z), 0.5f * (a.w + b.w));
        int k = q * 4;
        o0 += he.x * out_w[(k + 0) * 2] + he.y * out_w[(k + 1) * 2]
            + he.z * out_w[(k + 2) * 2] + he.w * out_w[(k + 3) * 2];
        o1 += he.x * out_w[(k + 0) * 2 + 1] + he.y * out_w[(k + 1) * 2 + 1]
            + he.z * out_w[(k + 2) * 2 + 1] + he.w * out_w[(k + 3) * 2 + 1];
    }
    float center = tanhf(o0);
    float span   = (50.0f / 50.0f) * tanhf(o1);
    float lo = center - 0.5f * span, hi = center + 0.5f * span;
    if (lo > hi) { float t = lo; lo = hi; hi = t; }
    pred[i * 2 + 0] = lo;
    pred[i * 2 + 1] = hi;
}

extern "C" void kernel_launch(void* const* d_in, const int* in_sizes, int n_in,
                              void* d_out, int out_size, void* d_ws, size_t ws_size,
                              hipStream_t stream) {
    const int*   edge_index = (const int*)  d_in[0];
    const int*   edge_type  = (const int*)  d_in[1];
    const float* edge_t     = (const float*)d_in[2];
    const int*   eids       = (const int*)  d_in[3];
    const float* ent_w      = (const float*)d_in[4];
    const float* rel_w      = (const float*)d_in[5];
    const float* tp_w       = (const float*)d_in[6];
    const float* tp_b       = (const float*)d_in[7];
    const float* g1_lin     = (const float*)d_in[8];
    const float* g1_lin_e   = (const float*)d_in[9];
    const float* g1_as      = (const float*)d_in[10];
    const float* g1_ad      = (const float*)d_in[11];
    const float* g1_ae      = (const float*)d_in[12];
    const float* g1_b       = (const float*)d_in[13];
    const float* g2_lin     = (const float*)d_in[14];
    const float* g2_lin_e   = (const float*)d_in[15];
    const float* g2_as      = (const float*)d_in[16];
    const float* g2_ad      = (const float*)d_in[17];
    const float* g2_ae      = (const float*)d_in[18];
    const float* g2_b       = (const float*)d_in[19];
    const float* out_w      = (const float*)d_in[20];
    const float* out_b      = (const float*)d_in[21];
    (void)in_sizes; (void)n_in; (void)out_size; (void)ws_size;

    size_t off = 0;
    auto alloc = [&](size_t bytes) -> void* {
        off = (off + 255) & ~(size_t)255;
        void* p = (char*)d_ws + off;
        off += bytes;
        return p;
    };
    unsigned short* wcatT = (unsigned short*)alloc((size_t)2 * 144 * 128 * sizeof(unsigned short));
    float* Me     = (float*)alloc((size_t)2 * 128 * 8 * sizeof(float));
    float* relAl  = (float*)alloc((size_t)2 * 500 * 8 * sizeof(float));
    float* TW     = (float*)alloc((size_t)2 * 64 * 8 * sizeof(float));
    float* tb     = (float*)alloc((size_t)2 * 8 * sizeof(float));
    float* loopAl = (float*)alloc((size_t)2 * 8 * sizeof(float));
    float* gsums  = (float*)alloc((size_t)80 * sizeof(float));
    float* alE    = (float*)alloc((size_t)2 * EE * 8 * sizeof(float));
    float* als    = (float*)alloc((size_t)NN * 8 * sizeof(float));
    float* ald    = (float*)alloc((size_t)NN * 8 * sizeof(float));
    float* xp     = (float*)alloc((size_t)NN * HIDD * sizeof(float));
    float* amax   = (float*)alloc((size_t)NN * 8 * sizeof(float));
    float* denom  = (float*)alloc((size_t)NN * 8 * sizeof(float));
    float* num    = (float*)alloc((size_t)NN * HIDD * sizeof(float));
    float* h1     = (float*)alloc((size_t)NN * HIDD * sizeof(float));

    hipMemsetAsync(gsums, 0, 80 * sizeof(float), stream);

    prep_wcat_kernel<<<dim3(304), dim3(128), 0, stream>>>(
        g1_lin, g1_lin_e, g1_as, g1_ad, g1_ae,
        g2_lin, g2_lin_e, g2_as, g2_ad, g2_ae, wcatT, Me);
    rel_tw_kernel<<<dim3((2 * 4520 + 255) / 256), dim3(256), 0, stream>>>(
        rel_w, tp_w, tp_b, Me, relAl, TW, tb);
    edge_al_kernel<<<dim3((EE + 255) / 256), dim3(256), 0, stream>>>(
        edge_t, edge_type, relAl, TW, tb, alE, gsums);
    loop_al_kernel<<<dim3(1), dim3(16), 0, stream>>>(gsums, TW, tb, loopAl);

    for (int L = 0; L < 2; ++L) {
        const float* x = L ? (const float*)h1 : ent_w;
        const unsigned short* wct = wcatT + (size_t)L * 144 * 128;
        const float* alE_L = alE + (size_t)L * EE * 8;
        const float* loopAl_L = loopAl + L * 8;
        gemm_xp_kernel<<<dim3((NN + 127) / 128), dim3(256), 0, stream>>>(x, wct, xp, als, ald);
        node_init_kernel<<<dim3((NN + 255) / 256), dim3(256), 0, stream>>>(als, ald, loopAl_L, amax);
        edge_max_kernel<<<dim3((EE + 255) / 256), dim3(256), 0, stream>>>(edge_index, als, ald, alE_L, amax);
        node_start_kernel<<<dim3((NN + 255) / 256), dim3(256), 0, stream>>>(als, ald, loopAl_L, amax, xp, denom, num);
        long long tot = (long long)EE * 32;
        edge_accum_kernel<<<dim3((unsigned)((tot + 255) / 256)), dim3(256), 0, stream>>>(
            edge_index, als, ald, alE_L, amax, xp, denom, num);
        node_fin_kernel<<<dim3((NN + 255) / 256), dim3(256), 0, stream>>>(
            num, denom, L ? g2_b : g1_b,
            L ? (const float*)h1 : (const float*)nullptr,
            L ? num : h1);
    }
    out_kernel<<<dim3((NEIDN + 255) / 256), dim3(256), 0, stream>>>(
        edge_index, eids, num, out_w, out_b, (float*)d_out);
}
